// AutoEncoder_14310831030501
// MI455X (gfx1250) — matrix operand order repaired and checked by reading, not compiled, not run on hardware
//
#include <hip/hip_runtime.h>
#include <hip/hip_bf16.h>
#include <math.h>

typedef __attribute__((ext_vector_type(16))) _Float16 v16h;
typedef __attribute__((ext_vector_type(4)))  _Float16 v4h;
typedef __attribute__((ext_vector_type(8)))  float    v8f;

#define NEG_INF (-__builtin_inff())

// K-permutation so each lane-half's A-fragment halfs are contiguous in LDS:
// c 0..7 -> 0..7, c 16..23 -> 8..15, c 8..15 -> 16..23, c 24..31 -> 24..31
__device__ __forceinline__ int pkperm(int c) {
  return ((c & 8) << 1) + ((c & 16) >> 1) + (c & 7);
}

// ---------------------------------------------------------------------------
// Generic WMMA GEMM: C[M,N] = act( A[M,K]@B[K,N] (+ A2[M,K2]@B2[K2,N]) + bias + res )
// Block tile 64x32, K-step 64, 8 waves (wave32), one 16x16 subtile per wave,
// 2x v_wmma_f32_16x16x32_f16 per LDS stage. A stored in natural k
// order, B stored transposed -> fragments load as 2x ds_load_b128 each.
// Requires M%64==0, N%32==0, K%64==0 (true for all call sites).
// ---------------------------------------------------------------------------
__global__ __launch_bounds__(256) void wmma_gemm_kernel(
    const float* __restrict__ A,  const float* __restrict__ B,
    const float* __restrict__ A2, const float* __restrict__ B2,
    const float* __restrict__ bias, const float* __restrict__ res,
    float* __restrict__ C, int M, int N, int K, int K2, int act)
{
  __shared__ __align__(16) _Float16 Ah[64 * 64];  // [row][kh*32 + c]
  __shared__ __align__(16) _Float16 Bt[32 * 64];  // [col][k]  (transposed)
  const int t    = threadIdx.x;
  const int lane = t & 31;
  const int wave = t >> 5;
  const int wm   = wave >> 1;          // 0..3  (M sub-tile)
  const int wn   = wave & 1;           // 0..1  (N sub-tile)
  const int row0 = blockIdx.x * 64;
  const int col0 = blockIdx.y * 32;
  const int m    = lane & 15;          // M row / N col within subtile
  const int hl   = lane >> 4;          // lane half

  v8f acc = {};

  const float* Ac = A; const float* Bc = B; int Kc = K;
  for (int pass = 0; pass < 2; ++pass) {
    if (pass == 1) { if (A2 == nullptr) break; Ac = A2; Bc = B2; Kc = K2; }
    for (int k0 = 0; k0 < Kc; k0 += 64) {
      // ---- stage A 64x64 tile: float4 global loads, row-major f16 LDS stores
      #pragma unroll
      for (int it = 0; it < 4; ++it) {
        int i  = t + it * 256;               // 1024 float4 total
        int r  = i >> 4, c4 = i & 15;        // c4: float4 index within row
        const float4 fv = *(const float4*)(Ac + (size_t)(row0 + r) * Kc + k0 + 4 * c4);
        int kh = c4 >> 3;                    // which 32-wide K half
        int cc = (c4 & 7) * 4;               // col within 32-block (4-aligned)
        v4h hv = { (_Float16)fv.x, (_Float16)fv.y, (_Float16)fv.z, (_Float16)fv.w };
        *(v4h*)(&Ah[r * 64 + kh * 32 + cc]) = hv;
      }
      // ---- stage B 64x32 tile: float4 global loads, transposed f16 stores
      #pragma unroll
      for (int it = 0; it < 2; ++it) {
        int i  = t + it * 256;               // 512 float4 total
        int r  = i >> 3;                     // K row 0..63
        int c4 = (i & 7) * 4;                // N col start
        const float4 fv = *(const float4*)(Bc + (size_t)(k0 + r) * N + col0 + c4);
        Bt[(c4 + 0) * 64 + r] = (_Float16)fv.x;
        Bt[(c4 + 1) * 64 + r] = (_Float16)fv.y;
        Bt[(c4 + 2) * 64 + r] = (_Float16)fv.z;
        Bt[(c4 + 3) * 64 + r] = (_Float16)fv.w;
      }
      __syncthreads();

      // ---- two WMMA steps (K halves), fragments are contiguous 32B loads
      const _Float16* Arow = &Ah[(wm * 16 + m) * 64 + hl * 16];
      const _Float16* Bcol = &Bt[(wn * 16 + m) * 64 + hl * 16];
      v16h a0 = *(const v16h*)(Arow);        // kh = 0
      v16h b0 = *(const v16h*)(Bcol);
      v16h a1 = *(const v16h*)(Arow + 32);   // kh = 1
      v16h b1 = *(const v16h*)(Bcol + 32);
      acc = __builtin_amdgcn_wmma_f32_16x16x32_f16(false, a0, false, b0,
                                                   (short)0, acc, false, false);
      acc = __builtin_amdgcn_wmma_f32_16x16x32_f16(false, a1, false, b1,
                                                   (short)0, acc, false, false);
      __syncthreads();
    }
  }

  // C layout: VGPR r -> M = 8*half + r ; N = lane&15
  const int col = col0 + wn * 16 + m;
  #pragma unroll
  for (int r = 0; r < 8; ++r) {
    int row = row0 + wm * 16 + 8 * hl + r;
    float v = acc[r];
    if (bias) v += bias[col];
    if (res)  v += res[(size_t)row * N + col];
    if (act == 1) v = fmaxf(v, 0.0f);
    C[(size_t)row * N + col] = v;
  }
}

// ---------------------------------------------------------------------------
// Utility kernels
// ---------------------------------------------------------------------------
__global__ void zero_kernel(float* p, long n) {
  long i = (long)blockIdx.x * blockDim.x + threadIdx.x;
  if (i < n) p[i] = 0.0f;
}
__global__ void copy_kernel(const float* __restrict__ a, float* __restrict__ b, long n) {
  long i = (long)blockIdx.x * blockDim.x + threadIdx.x;
  if (i < n) b[i] = a[i];
}
__global__ void count_kernel(const int* __restrict__ dst, float* cnt, int E) {
  int i = blockIdx.x * blockDim.x + threadIdx.x;
  if (i < E) atomicAdd(&cnt[dst[i]], 1.0f);
}
__global__ void scatter_kernel(const float* __restrict__ xp, const int* __restrict__ src,
                               const int* __restrict__ dst, float* mean, int E, int Dd) {
  long i = (long)blockIdx.x * blockDim.x + threadIdx.x;
  if (i >= (long)E * Dd) return;
  int e = (int)(i / Dd), c = (int)(i % Dd);
  atomicAdd(&mean[(size_t)dst[e] * Dd + c], xp[(size_t)src[e] * Dd + c]);
}
__global__ void divide_kernel(float* mean, const float* __restrict__ cnt, long n, int Dd) {
  long i = (long)blockIdx.x * blockDim.x + threadIdx.x;
  if (i < n) mean[i] /= fmaxf(cnt[i / Dd], 1.0f);
}
// L2 row-normalize in place; blockDim == W
__global__ void rownorm_kernel(float* x, int W) {
  extern __shared__ float sm[];
  int row = blockIdx.x, t = threadIdx.x;
  float v = x[(size_t)row * W + t];
  sm[t] = v * v; __syncthreads();
  for (int s = W >> 1; s > 0; s >>= 1) { if (t < s) sm[t] += sm[t + s]; __syncthreads(); }
  float nrm = sqrtf(sm[0]);
  x[(size_t)row * W + t] = v / fmaxf(nrm, 1e-12f);
}
// (optional silu) + layernorm; blockDim == W
__global__ void ln_kernel(const float* __restrict__ x, const float* __restrict__ g,
                          const float* __restrict__ b, float* __restrict__ y, int W, int silu) {
  extern __shared__ float sm[];
  int row = blockIdx.x, t = threadIdx.x;
  float v = x[(size_t)row * W + t];
  if (silu) v = v / (1.0f + __expf(-v));
  sm[t] = v; __syncthreads();
  for (int s = W >> 1; s > 0; s >>= 1) { if (t < s) sm[t] += sm[t + s]; __syncthreads(); }
  float mean = sm[0] / (float)W; __syncthreads();
  float d = v - mean;
  sm[t] = d * d; __syncthreads();
  for (int s = W >> 1; s > 0; s >>= 1) { if (t < s) sm[t] += sm[t + s]; __syncthreads(); }
  float var = sm[0] / (float)W;
  y[(size_t)row * W + t] = d * rsqrtf(var + 1e-5f) * g[t] + b[t];
}

// ---------------------------------------------------------------------------
// Fuser attention: each face attends to <=32 gathered edge keys (mask = loop).
// One 256-thread block per face. HEADS=2, hd=128, D=256. Duplicate loop
// entries are counted once (matches the boolean-mask reference semantics).
// ---------------------------------------------------------------------------
__global__ __launch_bounds__(256) void fuser_attn_kernel(
    const float* __restrict__ q, const float* __restrict__ k, const float* __restrict__ v,
    const int* __restrict__ loop, float* __restrict__ out)
{
  __shared__ __align__(16) float qs[256];
  __shared__ int   idx[32];
  __shared__ float sc[2][32];
  __shared__ float pr[2][32];
  __shared__ float hsum[2];
  const int face = blockIdx.x, t = threadIdx.x;
  qs[t] = q[(size_t)face * 256 + t];
  if (t < 32) {
    int e = loop[face * 32 + t];
    idx[t] = (e >= 0) ? e : -1;       // -1 / -2 => blocked
  }
  __syncthreads();
  if (t < 64) {
    int h = t >> 5, j = t & 31;
    int e = idx[j];
    bool dup = false;
    if (e >= 0) for (int jj = 0; jj < j; ++jj) if (idx[jj] == e) { dup = true; break; }
    float s = NEG_INF;
    if (e >= 0 && !dup) {
      s = 0.0f;
      const float4* kr = (const float4*)(k + (size_t)e * 256 + h * 128);
      const float4* qh = (const float4*)(qs + h * 128);
      #pragma unroll 8
      for (int d0 = 0; d0 < 32; ++d0) {
        float4 a = qh[d0], b = kr[d0];
        s += a.x * b.x + a.y * b.y + a.z * b.z + a.w * b.w;
      }
      s *= 0.0883883476483184f;       // 1/sqrt(128)
    }
    sc[h][j] = s;
  }
  __syncthreads();
  if (t < 2) {
    float mx = NEG_INF;
    for (int j = 0; j < 32; ++j) mx = fmaxf(mx, sc[t][j]);
    float sum = 0.0f;
    for (int j = 0; j < 32; ++j) {
      float s = sc[t][j];
      float p = (s == NEG_INF) ? 0.0f : __expf(s - mx);
      pr[t][j] = p; sum += p;
    }
    hsum[t] = sum;
  }
  __syncthreads();
  {
    int h = t >> 7, d0 = t & 127;
    float accv = 0.0f;
    for (int j = 0; j < 32; ++j) {
      float p = pr[h][j];
      int e = idx[j];
      if (e >= 0 && p > 0.0f) accv += p * v[(size_t)e * 256 + h * 128 + d0];
    }
    out[(size_t)face * 256 + t] = accv / fmaxf(hsum[h], 1e-30f);
  }
}

// ---------------------------------------------------------------------------
// Face self-attention with block-diagonal sample mask. One block per (query,head).
// Scores live in LDS (Lk <= 2048).
// ---------------------------------------------------------------------------
__global__ __launch_bounds__(256) void face_attn_kernel(
    const float* __restrict__ q, const float* __restrict__ k, const float* __restrict__ v,
    const int* __restrict__ sid, float* __restrict__ out, int Lk)
{
  __shared__ __align__(16) float qv[128];
  __shared__ float sc[2048];
  __shared__ float red[256];
  const int qi = blockIdx.x, h = blockIdx.y, t = threadIdx.x;
  const int lane = t & 31, wave = t >> 5;
  if (t < 128) qv[t] = q[(size_t)qi * 256 + h * 128 + t];
  __syncthreads();
  const int myid = sid[qi];
  const float4 qf = ((const float4*)qv)[lane];
  for (int j = wave; j < Lk; j += 8) {
    const float4 kf = *(const float4*)(k + (size_t)j * 256 + h * 128 + lane * 4);
    float part = qf.x * kf.x + qf.y * kf.y + qf.z * kf.z + qf.w * kf.w;
    for (int off = 16; off > 0; off >>= 1) part += __shfl_down(part, off, 32);
    if (lane == 0) sc[j] = (sid[j] == myid) ? part * 0.0883883476483184f : NEG_INF;
  }
  __syncthreads();
  float lm = NEG_INF;
  for (int j = t; j < Lk; j += 256) lm = fmaxf(lm, sc[j]);
  red[t] = lm; __syncthreads();
  for (int s = 128; s > 0; s >>= 1) { if (t < s) red[t] = fmaxf(red[t], red[t + s]); __syncthreads(); }
  float mx = red[0]; __syncthreads();
  float ls = 0.0f;
  for (int j = t; j < Lk; j += 256) {
    float s = sc[j];
    float p = (s == NEG_INF) ? 0.0f : __expf(s - mx);
    sc[j] = p; ls += p;
  }
  red[t] = ls; __syncthreads();
  for (int s = 128; s > 0; s >>= 1) { if (t < s) red[t] += red[t + s]; __syncthreads(); }
  float sum = red[0]; __syncthreads();
  if (t < 128) {
    float accv = 0.0f;
    for (int j = 0; j < Lk; ++j) accv += sc[j] * v[(size_t)j * 256 + h * 128 + t];
    out[(size_t)qi * 256 + h * 128 + t] = accv / fmaxf(sum, 1e-30f);
  }
}

// ---------------------------------------------------------------------------
// Host orchestration
// ---------------------------------------------------------------------------
extern "C" void kernel_launch(void* const* d_in, const int* in_sizes, int n_in,
                              void* d_out, int out_size, void* d_ws, size_t ws_size,
                              hipStream_t stream) {
  constexpr int S = 8192, L = 2048, D = 256, E = 131072;
  const float* edge_emb = (const float*)d_in[0];
  const float* face_emb = (const float*)d_in[1];
  const int*   adj      = (const int*)d_in[2];
  const int*   loop     = (const int*)d_in[3];
  const int*   fsid     = (const int*)d_in[4];
  auto F = [&](int i) { return (const float*)d_in[i]; };
  const int* src = adj;
  const int* dst = adj + E;

  // workspace layout (floats): bx | by | xp | mn | cnt
  float* ws  = (float*)d_ws;
  float* bx  = ws;
  float* by  = ws + (size_t)S * D;
  float* xp  = ws + 2 * (size_t)S * D;
  float* mn  = ws + 3 * (size_t)S * D;
  float* cnt = ws + 4 * (size_t)S * D;
  float* ep   = (float*)d_out;                 // edge_plus output [S,D]
  float* fout = ep + (size_t)S * D;            // f output [L,D]
  // after sage chain, reuse:
  float* qb  = bx;
  float* att = bx + (size_t)L * D;
  float* f   = bx + 2 * (size_t)L * D;
  float* f2  = bx + 3 * (size_t)L * D;
  float* kb  = xp;
  float* vb  = mn;

  auto gemm = [&](const float* A, const float* B, const float* A2, const float* B2,
                  const float* bias, const float* res, float* C,
                  int M, int N, int K, int K2, int act) {
    dim3 g(M / 64, N / 32);
    wmma_gemm_kernel<<<g, 256, 0, stream>>>(A, B, A2, B2, bias, res, C, M, N, K, K2, act);
  };
  auto zero = [&](float* p, long n) {
    zero_kernel<<<(unsigned)((n + 255) / 256), 256, 0, stream>>>(p, n);
  };

  // degree counts (shared by all SAGE layers)
  zero(cnt, S);
  count_kernel<<<(E + 255) / 256, 256, 0, stream>>>(dst, cnt, E);

  // ---- SAGE chain: edge_emb -> bx -> (silu+ln0) by -> bx -> by -> ep ----
  const int sdi[4] = {256, 64, 128, 256};
  const int sdo[4] = {64, 128, 256, 256};
  float* outs[4] = { bx, bx, by, ep };
  const float* xin = edge_emb;
  for (int i = 0; i < 4; ++i) {
    int di = sdi[i], doo = sdo[i];
    int pb = 5 + i * 5;  // pw, pb, lw, lb, rw
    const float *pw = F(pb), *pbb = F(pb + 1), *lw = F(pb + 2), *lb = F(pb + 3), *rw = F(pb + 4);
    // project: xp = relu(x @ pw + pb)
    gemm(xin, pw, nullptr, nullptr, pbb, nullptr, xp, S, di, di, 0, 1);
    // mean aggregation over edges
    zero(mn, (long)S * di);
    scatter_kernel<<<(unsigned)(((long)E * di + 255) / 256), 256, 0, stream>>>(xp, src, dst, mn, E, di);
    divide_kernel<<<(unsigned)(((long)S * di + 255) / 256), 256, 0, stream>>>(mn, cnt, (long)S * di, di);
    // out = mean@lw + x@rw + lb ; then L2-normalize rows
    float* out = outs[i];
    gemm(mn, lw, xin, rw, lb, nullptr, out, S, doo, di, di, 0);
    rownorm_kernel<<<S, doo, doo * sizeof(float), stream>>>(out, doo);
    if (i == 0) {
      ln_kernel<<<S, 64, 64 * sizeof(float), stream>>>(out, F(25), F(26), by, 64, /*silu=*/1);
      xin = by;
    } else {
      xin = out;
    }
  }
  // sage3 output (edge_plus) already in d_out[0 : S*D]

  // ---- Attn_fuser: f = LN(f + MHA(f, edge_plus, loop-mask)) x2 ----
  copy_kernel<<<(unsigned)(((long)L * D + 255) / 256), 256, 0, stream>>>(face_emb, f, (long)L * D);
  for (int l = 0; l < 2; ++l) {
    int b0 = 27 + l * 10;  // wq,wk,wv,bq,bk,bv,wo,bo,lng,lnb
    gemm(f,  F(b0 + 0), nullptr, nullptr, F(b0 + 3), nullptr, qb, L, D, D, 0, 0);
    gemm(ep, F(b0 + 1), nullptr, nullptr, F(b0 + 4), nullptr, kb, S, D, D, 0, 0);
    gemm(ep, F(b0 + 2), nullptr, nullptr, F(b0 + 5), nullptr, vb, S, D, D, 0, 0);
    fuser_attn_kernel<<<L, 256, 0, stream>>>(qb, kb, vb, loop, att);
    gemm(att, F(b0 + 6), nullptr, nullptr, F(b0 + 7), f, f2, L, D, D, 0, 0);  // +residual
    ln_kernel<<<L, 256, 256 * sizeof(float), stream>>>(f2, F(b0 + 8), F(b0 + 9), f, 256, 0);
  }

  // ---- Face_atten: f = LN(f + MHA(f, f, block-diagonal)) x2 ----
  for (int l = 0; l < 2; ++l) {
    int b0 = 47 + l * 10;
    gemm(f, F(b0 + 0), nullptr, nullptr, F(b0 + 3), nullptr, qb, L, D, D, 0, 0);
    gemm(f, F(b0 + 1), nullptr, nullptr, F(b0 + 4), nullptr, kb, L, D, D, 0, 0);
    gemm(f, F(b0 + 2), nullptr, nullptr, F(b0 + 5), nullptr, vb, L, D, D, 0, 0);
    face_attn_kernel<<<dim3(L, 2), 256, 0, stream>>>(qb, kb, vb, fsid, att, L);
    gemm(att, F(b0 + 6), nullptr, nullptr, F(b0 + 7), f, f2, L, D, D, 0, 0);
    ln_kernel<<<L, 256, 256 * sizeof(float), stream>>>(f2, F(b0 + 8), F(b0 + 9), f, 256, 0);
  }

  // final f -> d_out tail
  copy_kernel<<<(unsigned)(((long)L * D + 255) / 256), 256, 0, stream>>>(f, fout, (long)L * D);
}
